// GAT_63720134804085
// MI455X (gfx1250) — matrix operand order repaired and checked by reading, not compiled, not run on hardware
//
#include <hip/hip_runtime.h>
#include <hip/hip_bf16.h>
#include <cstdint>
#include <cstddef>

// Problem constants (match reference)
#define NN        4096
#define FIN       512
#define FHID      128
#define NHEADS    4
#define FOUT2     64
#define LRELU_A   0.2f
#define MASK_FILL -9.0e15f

typedef __attribute__((ext_vector_type(16))) __bf16 v16bf;
typedef __attribute__((ext_vector_type(8)))  __bf16 v8bf;
typedef __attribute__((ext_vector_type(8)))  float  v8f;

__device__ __forceinline__ __bf16 f2bf(float f) {
  unsigned u = __builtin_bit_cast(unsigned, f);
  unsigned r = u + 0x7FFFu + ((u >> 16) & 1u);   // round-to-nearest-even
  unsigned short h = (unsigned short)(r >> 16);
  return __builtin_bit_cast(__bf16, h);
}

__device__ __forceinline__ v8f wmma_bf16f32(v16bf a, v16bf b, v8f c) {
#if defined(__HIP_DEVICE_COMPILE__)
  return __builtin_amdgcn_wmma_f32_16x16x32_bf16(false, a, false, b, (short)0, c,
                                                 false, false);
#else
  return c;
#endif
}

// ---------------- elementwise f32 -> bf16 ----------------
__global__ void k_cvt(const float* __restrict__ src, __bf16* __restrict__ dst, int n) {
  int i = blockIdx.x * blockDim.x + threadIdx.x;
  if (i < n) dst[i] = f2bf(src[i]);
}

// ---------------- transpose + convert: src f32 [z][R][C] -> dst bf16 [z][C][R] ----
__global__ void k_convT(const float* __restrict__ src, __bf16* __restrict__ dst,
                        int R, int C) {
  size_t base = (size_t)blockIdx.z * (size_t)R * (size_t)C;
  int idx = blockIdx.x * blockDim.x + threadIdx.x;
  if (idx >= R * C) return;
  int r = idx / C, c = idx - r * C;
  dst[base + (size_t)c * R + r] = f2bf(src[base + (size_t)r * C + c]);
}

// ---------------- attention score vectors: s1=Wh@a[:F], s2=Wh@a[F:] -------------
// block = 256 (8 waves), each wave reduces one row. grid.z = head.
__global__ void k_scores(const float* __restrict__ Wh, const float* __restrict__ a,
                         float* __restrict__ s1, float* __restrict__ s2,
                         int F, size_t whStride, size_t aStride) {
  int lane = threadIdx.x & 31;
  int wave = threadIdx.x >> 5;
  int z = blockIdx.z;
  int row = blockIdx.x * 8 + wave;
  const float* wr = Wh + (size_t)z * whStride + (size_t)row * F;
  const float* az = a + (size_t)z * aStride;
  float acc1 = 0.f, acc2 = 0.f;
  for (int f = lane; f < F; f += 32) {
    float v = wr[f];
    acc1 += v * az[f];
    acc2 += v * az[F + f];
  }
  #pragma unroll
  for (int o = 16; o > 0; o >>= 1) {
    acc1 += __shfl_xor(acc1, o, 32);
    acc2 += __shfl_xor(acc2, o, 32);
  }
  if (lane == 0) {
    s1[(size_t)z * NN + row] = acc1;
    s2[(size_t)z * NN + row] = acc2;
  }
}

// ---------------- bf16 WMMA GEMM: C[M,Nc] = A[M,K] @ B[K,Nc] --------------------
// A row-major bf16, BT row-major [Nc][K] bf16 (i.e. B column-major), C fp32.
// block = 128 (4 waves); wave -> one 16x64 slab (4 wmma accumulators).
__global__ void k_gemm_bf16(const __bf16* __restrict__ A,
                            const __bf16* __restrict__ BT,
                            float* __restrict__ C,
                            int K, int Nc, size_t strideB, size_t strideC) {
  int lane = threadIdx.x & 31;
  int wave = threadIdx.x >> 5;
  int half = lane >> 4;
  int l15  = lane & 15;
  int m0 = (blockIdx.x * 4 + wave) * 16;
  int n0 = blockIdx.y * 64;
  const __bf16* Bz = BT + (size_t)blockIdx.z * strideB;
  float* Cz = C + (size_t)blockIdx.z * strideC;

  v8f acc[4];
  #pragma unroll
  for (int t = 0; t < 4; ++t) {
    #pragma unroll
    for (int r = 0; r < 8; ++r) acc[t][r] = 0.f;
  }

  for (int k0 = 0; k0 < K; k0 += 32) {
    // A fragment: lane half0 -> K 0..15; half1 -> K 16..31
    const __bf16* ap = A + (size_t)(m0 + l15) * K + k0 + half * 16;
    v8bf alo = *(const v8bf*)ap;
    v8bf ahi = *(const v8bf*)(ap + 8);
    v16bf a;
    #pragma unroll
    for (int i = 0; i < 8; ++i) { a[i] = alo[i]; a[i + 8] = ahi[i]; }
    #pragma unroll
    for (int t = 0; t < 4; ++t) {
      // B fragment: lane = column, lanes<16 K 0..15, lanes>=16 K 16..31
      const __bf16* bp = Bz + (size_t)(n0 + t * 16 + l15) * K + k0 + half * 16;
      v16bf b = *(const v16bf*)bp;
      acc[t] = wmma_bf16f32(a, b, acc[t]);
    }
  }
  #pragma unroll
  for (int t = 0; t < 4; ++t) {
    #pragma unroll
    for (int r = 0; r < 8; ++r)
      Cz[(size_t)(m0 + r + half * 8) * Nc + n0 + t * 16 + l15] = acc[t][r];
  }
}

// ---------------- flash-style masked-softmax aggregation ------------------------
// out[i,:] = softmax_j( mask ? leakyrelu(s1[i]+s2[j]) : -9e15 ) @ Wh
// WhT: bf16 [FOUT][NN] per head. One wave per 16-row block; block = 128.
template <int TILES>
__global__ void k_flash(const int* __restrict__ g,
                        const float* __restrict__ s1,
                        const float* __restrict__ s2,
                        const __bf16* __restrict__ WhT,
                        float* __restrict__ out,
                        int ldo, int colPerHead,
                        size_t sStride, size_t whtStride, int elu) {
  int lane = threadIdx.x & 31;
  int wave = threadIdx.x >> 5;
  int half = lane >> 4;
  int l15  = lane & 15;
  int z = blockIdx.z;
  int rb = blockIdx.x * 4 + wave;
  int row = rb * 16 + l15;

  const float*  s1z = s1 + (size_t)z * sStride;
  const float*  s2z = s2 + (size_t)z * sStride;
  const __bf16* whz = WhT + (size_t)z * whtStride;
  const int*    grow = g + (size_t)row * NN;

  float s1v = s1z[row];
  float m = -__builtin_inff();
  float l = 0.f;
  v8f acc[TILES];
  #pragma unroll
  for (int t = 0; t < TILES; ++t) {
    #pragma unroll
    for (int r = 0; r < 8; ++r) acc[t][r] = 0.f;
  }

  for (int j0 = 0; j0 < NN; j0 += 32) {
    int c0 = j0 + half * 16;       // cols c0..c0+7  (A-frag elems 0..7)
    int c1 = c0 + 8;               // cols c1..c1+7  (A-frag elems 8..15)
    __attribute__((aligned(16))) int   gv[16];
    __attribute__((aligned(16))) float s2v[16];
    *(int4*)(gv + 0)     = *(const int4*)(grow + c0);
    *(int4*)(gv + 4)     = *(const int4*)(grow + c0 + 4);
    *(int4*)(gv + 8)     = *(const int4*)(grow + c1);
    *(int4*)(gv + 12)    = *(const int4*)(grow + c1 + 4);
    *(float4*)(s2v + 0)  = *(const float4*)(s2z + c0);
    *(float4*)(s2v + 4)  = *(const float4*)(s2z + c0 + 4);
    *(float4*)(s2v + 8)  = *(const float4*)(s2z + c1);
    *(float4*)(s2v + 12) = *(const float4*)(s2z + c1 + 4);

    float e[16];
    float tmax = -__builtin_inff();
    #pragma unroll
    for (int i = 0; i < 16; ++i) {
      float x = s1v + s2v[i];
      x = x > 0.f ? x : LRELU_A * x;
      e[i] = (gv[i] > 0) ? x : MASK_FILL;     // finite fill == reference semantics
      tmax = fmaxf(tmax, e[i]);
    }
    tmax = fmaxf(tmax, __shfl_xor(tmax, 16, 32));
    float newm = fmaxf(m, tmax);
    float scl = __expf(m - newm);              // first iter: exp(-inf)=0

    float p[16];
    float rs = 0.f;
    #pragma unroll
    for (int i = 0; i < 16; ++i) { p[i] = __expf(e[i] - newm); rs += p[i]; }
    rs += __shfl_xor(rs, 16, 32);
    l = l * scl + rs;
    m = newm;

    // per-row rescale of accumulators (row r+8*half stats live in lane r+8*half)
    float srow[8];
    #pragma unroll
    for (int r = 0; r < 8; ++r) srow[r] = __shfl(scl, r + half * 8, 32);
    #pragma unroll
    for (int t = 0; t < TILES; ++t) {
      #pragma unroll
      for (int r = 0; r < 8; ++r) acc[t][r] *= srow[r];
    }

    // probabilities as bf16 A-fragment
    v16bf pa;
    #pragma unroll
    for (int i = 0; i < 8; ++i) { pa[i] = f2bf(p[i]); pa[i + 8] = f2bf(p[i + 8]); }

    #pragma unroll
    for (int t = 0; t < TILES; ++t) {
      const __bf16* bp = whz + (size_t)(t * 16 + l15) * NN + j0 + half * 16;
      v16bf b = *(const v16bf*)bp;
      acc[t] = wmma_bf16f32(pa, b, acc[t]);
    }
  }

  float lrow[8];
  #pragma unroll
  for (int r = 0; r < 8; ++r) lrow[r] = __shfl(l, r + half * 8, 32);
  int cb = colPerHead * z;
  #pragma unroll
  for (int t = 0; t < TILES; ++t) {
    #pragma unroll
    for (int r = 0; r < 8; ++r) {
      float o = acc[t][r] / lrow[r];
      if (elu) o = o > 0.f ? o : (__expf(o) - 1.f);
      out[(size_t)(rb * 16 + r + half * 8) * ldo + cb + t * 16 + l15] = o;
    }
  }
}

// ---------------------------------------------------------------------------------
extern "C" void kernel_launch(void* const* d_in, const int* in_sizes, int n_in,
                              void* d_out, int out_size, void* d_ws, size_t ws_size,
                              hipStream_t stream) {
  const int*   g  = (const int*)d_in[0];
  const float* X  = (const float*)d_in[1];
  const float* W1 = (const float*)d_in[2];
  const float* a1 = (const float*)d_in[3];
  const float* W2 = (const float*)d_in[4];
  const float* a2 = (const float*)d_in[5];
  float* out = (float*)d_out;

  char* ws = (char*)d_ws;
  size_t off = 0;
  auto take = [&](size_t bytes) -> char* {
    char* p = ws + off;
    off += (bytes + 255) & ~(size_t)255;
    return p;
  };

  __bf16* Xbf  = (__bf16*)take((size_t)NN * FIN * 2);
  __bf16* W1T  = (__bf16*)take((size_t)NHEADS * FHID * FIN * 2);
  float*  Wh1  = (float*) take((size_t)NHEADS * NN * FHID * 4);
  __bf16* WhT1 = (__bf16*)take((size_t)NHEADS * FHID * NN * 2);
  float*  s1a  = (float*) take((size_t)NHEADS * NN * 4);
  float*  s2a  = (float*) take((size_t)NHEADS * NN * 4);
  float*  hbuf = (float*) take((size_t)NN * (FHID * NHEADS) * 4);
  __bf16* hbf  = (__bf16*)take((size_t)NN * (FHID * NHEADS) * 2);
  __bf16* W2T  = (__bf16*)take((size_t)FOUT2 * (FHID * NHEADS) * 2);
  float*  Wh2  = (float*) take((size_t)NN * FOUT2 * 4);
  __bf16* WhT2 = (__bf16*)take((size_t)FOUT2 * NN * 2);
  float*  s1b  = (float*) take((size_t)NN * 4);
  float*  s2b  = (float*) take((size_t)NN * 4);
  (void)ws_size; (void)in_sizes; (void)n_in; (void)out_size;

  // ---- layer 1 ----
  k_cvt<<<dim3((NN * FIN + 255) / 256), 256, 0, stream>>>(X, Xbf, NN * FIN);
  k_convT<<<dim3((FIN * FHID + 255) / 256, 1, NHEADS), 256, 0, stream>>>(W1, W1T, FIN, FHID);
  k_gemm_bf16<<<dim3(NN / 64, FHID / 64, NHEADS), 128, 0, stream>>>(
      Xbf, W1T, Wh1, FIN, FHID, (size_t)FHID * FIN, (size_t)NN * FHID);
  k_scores<<<dim3(NN / 8, 1, NHEADS), 256, 0, stream>>>(
      Wh1, a1, s1a, s2a, FHID, (size_t)NN * FHID, (size_t)2 * FHID);
  k_convT<<<dim3((NN * FHID + 255) / 256, 1, NHEADS), 256, 0, stream>>>(Wh1, WhT1, NN, FHID);
  k_flash<8><<<dim3(NN / 64, 1, NHEADS), 128, 0, stream>>>(
      g, s1a, s2a, WhT1, hbuf, FHID * NHEADS, FHID,
      (size_t)NN, (size_t)FHID * NN, 1);

  // ---- layer 2 ----
  k_cvt<<<dim3((NN * FHID * NHEADS + 255) / 256), 256, 0, stream>>>(hbuf, hbf, NN * FHID * NHEADS);
  k_convT<<<dim3((FHID * NHEADS * FOUT2 + 255) / 256, 1, 1), 256, 0, stream>>>(
      W2, W2T, FHID * NHEADS, FOUT2);
  k_gemm_bf16<<<dim3(NN / 64, 1, 1), 128, 0, stream>>>(
      hbf, W2T, Wh2, FHID * NHEADS, FOUT2, 0, 0);
  k_scores<<<dim3(NN / 8, 1, 1), 256, 0, stream>>>(
      Wh2, a2, s1b, s2b, FOUT2, 0, 0);
  k_convT<<<dim3((NN * FOUT2 + 255) / 256, 1, 1), 256, 0, stream>>>(Wh2, WhT2, NN, FOUT2);
  k_flash<4><<<dim3(NN / 64, 1, 1), 128, 0, stream>>>(
      g, s1b, s2b, WhT2, out, FOUT2, 0, 0, 0, 0);
}
